// Attention_15857019256892
// MI455X (gfx1250) — matrix operand order repaired and checked by reading, not compiled, not run on hardware
//
#include <hip/hip_runtime.h>
#include <hip/hip_bf16.h>

// ---------------------------------------------------------------------------
// Types for CDNA5 WMMA (wave32)
// ---------------------------------------------------------------------------
typedef __attribute__((ext_vector_type(16))) __bf16        v16bf;
typedef __attribute__((ext_vector_type(4)))  __bf16        v4bf;
typedef __attribute__((ext_vector_type(8)))  float         v8f;
typedef __attribute__((ext_vector_type(4)))  unsigned int  v4u;

#define D_MODEL 2048
#define SEQ     2048
#define NBATCH  2

union FragU { v4u u[2]; v16bf v; };

__device__ __forceinline__ v8f wmma_bf16(v16bf a, v16bf b, v8f c) {
  // D = A(16x32 bf16) * B(32x16 bf16) + C(16x16 f32)
  return __builtin_amdgcn_wmma_f32_16x16x32_bf16(
      /*neg_a=*/false, a, /*neg_b=*/false, b,
      /*c_mod=*/(short)0, c, /*reuse_a=*/false, /*reuse_b=*/false);
}

// Async copy of 16 bytes: global -> LDS, tracked by ASYNCcnt.
// lds_off = LDS byte address (low 32 bits of flat shared-aperture address).
__device__ __forceinline__ void async_copy_b128(unsigned lds_off, const void* gaddr) {
  asm volatile("global_load_async_to_lds_b128 %0, %1, off"
               :: "v"(lds_off), "v"(gaddr) : "memory");
}

__device__ __forceinline__ unsigned lds_addr_of(const void* p) {
  return (unsigned)(unsigned long long)p;   // LDS aperture: offset = addr[31:0]
}

// ---------------------------------------------------------------------------
// fp32 -> bf16 flat conversion (n divisible by 4)
// ---------------------------------------------------------------------------
__global__ __launch_bounds__(256) void cvt_bf16_kernel(
    const float* __restrict__ in, __bf16* __restrict__ out, int n) {
  int idx = (blockIdx.x * blockDim.x + threadIdx.x) * 4;
  if (idx >= n) return;
  float4 f = *(const float4*)(in + idx);
  v4bf o;
  o[0] = (__bf16)f.x; o[1] = (__bf16)f.y;
  o[2] = (__bf16)f.z; o[3] = (__bf16)f.w;
  *(v4bf*)(out + idx) = o;
}

// ---------------------------------------------------------------------------
// fp32 (dim x dim) -> bf16 transposed (out[n][k] = in[k][n]); LDS-tiled
// ---------------------------------------------------------------------------
__global__ __launch_bounds__(256) void transpose_cvt_kernel(
    const float* __restrict__ in, __bf16* __restrict__ out, int dim) {
  __shared__ float tile[32][33];
  int bx = blockIdx.x * 32, by = blockIdx.y * 32;
  int tx = threadIdx.x, ty = threadIdx.y;       // block = (32, 8)
  #pragma unroll
  for (int i = 0; i < 32; i += 8)
    tile[ty + i][tx] = in[(long)(by + ty + i) * dim + (bx + tx)];
  __syncthreads();
  #pragma unroll
  for (int i = 0; i < 32; i += 8)
    out[(long)(bx + ty + i) * dim + (by + tx)] = (__bf16)tile[tx][ty + i];
}

// ---------------------------------------------------------------------------
// Row softmax with scale: attn = softmax(scores * scale), bf16 output.
// One 256-thread block (8 waves) per row of length SEQ.
// ---------------------------------------------------------------------------
__global__ __launch_bounds__(256) void softmax_scale_kernel(
    const float* __restrict__ scores, __bf16* __restrict__ attn, float scale) {
  __shared__ float smax[8];
  __shared__ float ssum[8];
  const long row = blockIdx.x;
  const float* src = scores + row * (long)SEQ;
  __bf16* dst = attn + row * (long)SEQ;
  const int t = threadIdx.x;
  const int lane = t & 31, wid = t >> 5;

  float v[8];
  float4 x0 = *(const float4*)(src + t * 8);
  float4 x1 = *(const float4*)(src + t * 8 + 4);
  v[0] = x0.x * scale; v[1] = x0.y * scale; v[2] = x0.z * scale; v[3] = x0.w * scale;
  v[4] = x1.x * scale; v[5] = x1.y * scale; v[6] = x1.z * scale; v[7] = x1.w * scale;

  float mx = v[0];
  #pragma unroll
  for (int i = 1; i < 8; ++i) mx = fmaxf(mx, v[i]);
  #pragma unroll
  for (int off = 16; off > 0; off >>= 1) mx = fmaxf(mx, __shfl_xor(mx, off, 32));
  if (lane == 0) smax[wid] = mx;
  __syncthreads();
  float bm = smax[0];
  #pragma unroll
  for (int i = 1; i < 8; ++i) bm = fmaxf(bm, smax[i]);

  float s = 0.f;
  #pragma unroll
  for (int i = 0; i < 8; ++i) { v[i] = __expf(v[i] - bm); s += v[i]; }
  #pragma unroll
  for (int off = 16; off > 0; off >>= 1) s += __shfl_xor(s, off, 32);
  if (lane == 0) ssum[wid] = s;
  __syncthreads();
  float bs = 0.f;
  #pragma unroll
  for (int i = 0; i < 8; ++i) bs += ssum[i];
  float inv = 1.f / bs;

  v4bf o0, o1;
  #pragma unroll
  for (int i = 0; i < 4; ++i) o0[i] = (__bf16)(v[i] * inv);
  #pragma unroll
  for (int i = 0; i < 4; ++i) o1[i] = (__bf16)(v[i + 4] * inv);
  *(v4bf*)(dst + t * 8)     = o0;
  *(v4bf*)(dst + t * 8 + 4) = o1;
}

// ---------------------------------------------------------------------------
// C-tile store epilogues
// Store modes: 0 = fp32 row-major, 1 = bf16 row-major,
//              2 = bf16 transposed per batch (for V^T): out[b][n][s]
// ---------------------------------------------------------------------------
template<int MODE>
__device__ __forceinline__ void store_tile(const v8f& acc, int m0, int n0,
                                           void* Cv, long cOff, int ldc,
                                           int h, int r) {
  const int n = n0 + r;
  if (MODE == 0) {
    float* Cf = (float*)Cv + cOff;
    #pragma unroll
    for (int i = 0; i < 8; ++i) {
      int m = m0 + h * 8 + i;
      Cf[(long)m * ldc + n] = acc[i];
    }
  } else if (MODE == 1) {
    __bf16* Cb = (__bf16*)Cv + cOff;
    #pragma unroll
    for (int i = 0; i < 8; ++i) {
      int m = m0 + h * 8 + i;
      Cb[(long)m * ldc + n] = (__bf16)acc[i];
    }
  } else {
    // m is a global token index (b*SEQ + s); store Vt[b][n][s]
    __bf16* Cb = (__bf16*)Cv;
    #pragma unroll
    for (int i = 0; i < 8; ++i) {
      int m = m0 + h * 8 + i;
      long off = ((long)(m >> 11) * D_MODEL + n) * (long)SEQ + (m & (SEQ - 1));
      Cb[off] = (__bf16)acc[i];
    }
  }
}

// ---------------------------------------------------------------------------
// bf16 WMMA GEMM with async global->LDS double buffering:
//   C(MxN) = A(MxK, row-major) * Bt(NxK, row-major)^T
// Block = 256 threads (8 waves); wave = 32x32 C; block tile = 64x128;
// K-step 32, K-loop unrolled x2 so the LDS buffer index is a compile-time
// constant (branch-free steady state). Tiles staged via
// GLOBAL_LOAD_ASYNC_TO_LDS_B128 (3 async ops / thread / tile, waited with
// s_wait_asynccnt 3 so tile i+1's copy overlaps tile i's WMMAs). The final
// trailing issue wraps to k=0 (harmless refetch into the dead buffer) to
// keep the loop body unconditional. Requires K % 64 == 0.
// ---------------------------------------------------------------------------
#define KSTEP 32
#define LDT   40   // padded LDS row stride in elements (80 B, 16B-aligned)

template<int MODE>
__global__ __launch_bounds__(256) void gemm_bf16_wmma(
    const __bf16* __restrict__ A, const __bf16* __restrict__ Bt,
    void* __restrict__ C, int N, int K,
    long strideA, long strideB, long strideC) {
  __shared__ __bf16 tA[2][64  * LDT];
  __shared__ __bf16 tB[2][128 * LDT];
  constexpr unsigned ABYTES = 64  * LDT * 2;
  constexpr unsigned BBYTES = 128 * LDT * 2;

  const int t    = threadIdx.x;
  const int lane = t & 31;
  const int wid  = t >> 5;
  const int wm   = wid >> 2;       // 0..1
  const int wn   = wid & 3;        // 0..3
  const long zb  = blockIdx.z;

  const __bf16* Ab = A  + zb * strideA;
  const __bf16* Bb = Bt + zb * strideB;

  const int mBlk = blockIdx.y * 64;
  const int nBlk = blockIdx.x * 128;

  const int r = lane & 15;   // row within A-tile / col within B-tile
  const int h = lane >> 4;   // K-half selector

  // --- cooperative copy coordinates ---------------------------------------
  const int arow = t >> 2, acol = (t & 3) * 8;    // A: 64 rows x 32 elems
  const int brow = t >> 1, bcol = (t & 1) * 16;   // B: 128 rows x 32 elems
  const __bf16* gA = Ab + (long)(mBlk + arow) * K + acol;
  const __bf16* gB = Bb + (long)(nBlk + brow) * K + bcol;
  const unsigned lA = lds_addr_of(&tA[0][arow * LDT + acol]);
  const unsigned lB = lds_addr_of(&tB[0][brow * LDT + bcol]);

  auto issue_tile = [&](int k0, int buf) {
    async_copy_b128(lA + (unsigned)buf * ABYTES,      gA + k0);
    async_copy_b128(lB + (unsigned)buf * BBYTES,      gB + k0);
    async_copy_b128(lB + (unsigned)buf * BBYTES + 16, gB + k0 + 8);
  };

  v8f acc00 = {}, acc01 = {}, acc10 = {}, acc11 = {};

  auto compute_step = [&](int buf) {   // buf is a compile-time constant at use
    const __bf16* sA = &tA[buf][0];
    const __bf16* sB = &tB[buf][0];
    FragU fa0, fa1, fb0, fb1;
    fa0.u[0] = *(const v4u*)(sA + (wm * 32 + r)      * LDT + h * 8);
    fa0.u[1] = *(const v4u*)(sA + (wm * 32 + r)      * LDT + 16 + h * 8);
    fa1.u[0] = *(const v4u*)(sA + (wm * 32 + 16 + r) * LDT + h * 8);
    fa1.u[1] = *(const v4u*)(sA + (wm * 32 + 16 + r) * LDT + 16 + h * 8);
    fb0.u[0] = *(const v4u*)(sB + (wn * 32 + r)      * LDT + h * 8);
    fb0.u[1] = *(const v4u*)(sB + (wn * 32 + r)      * LDT + 16 + h * 8);
    fb1.u[0] = *(const v4u*)(sB + (wn * 32 + 16 + r) * LDT + h * 8);
    fb1.u[1] = *(const v4u*)(sB + (wn * 32 + 16 + r) * LDT + 16 + h * 8);

    acc00 = wmma_bf16(fa0.v, fb0.v, acc00);
    acc01 = wmma_bf16(fa0.v, fb1.v, acc01);
    acc10 = wmma_bf16(fa1.v, fb0.v, acc10);
    acc11 = wmma_bf16(fa1.v, fb1.v, acc11);
  };

  issue_tile(0, 0);
  for (int k0 = 0; k0 < K; k0 += 2 * KSTEP) {
    const int k1 = k0 + KSTEP;                       // always < K (K % 64 == 0)
    int k2 = k1 + KSTEP; if (k2 >= K) k2 = 0;        // wrap: dead-buffer refetch

    issue_tile(k1, 1);
    asm volatile("s_wait_asynccnt 0x3" ::: "memory");  // tile(buf0) copies done
    __syncthreads();
    compute_step(0);
    __syncthreads();

    issue_tile(k2, 0);
    asm volatile("s_wait_asynccnt 0x3" ::: "memory");  // tile(buf1) copies done
    __syncthreads();
    compute_step(1);
    __syncthreads();
  }

  const long cOff = zb * strideC;
  const int m0 = mBlk + wm * 32;
  const int n0 = nBlk + wn * 32;
  store_tile<MODE>(acc00, m0,      n0,      C, cOff, N, h, r);
  store_tile<MODE>(acc01, m0,      n0 + 16, C, cOff, N, h, r);
  store_tile<MODE>(acc10, m0 + 16, n0,      C, cOff, N, h, r);
  store_tile<MODE>(acc11, m0 + 16, n0 + 16, C, cOff, N, h, r);
}

// ---------------------------------------------------------------------------
// Host-side launch
// ---------------------------------------------------------------------------
extern "C" void kernel_launch(void* const* d_in, const int* in_sizes, int n_in,
                              void* d_out, int out_size, void* d_ws, size_t ws_size,
                              hipStream_t stream) {
  const int D = D_MODEL, S = SEQ, B = NBATCH;
  const int M = B * S;                       // 4096 tokens
  const float scale = 1.0f / sqrtf((float)D);

  const float* x  = (const float*)d_in[0];
  const float* Wq = (const float*)d_in[1];
  const float* Wk = (const float*)d_in[2];
  const float* Wv = (const float*)d_in[3];
  const float* Wo = (const float*)d_in[4];

  // Workspace carve-out (~168 MB total)
  char* ws = (char*)d_ws;
  size_t off = 0;
  auto carve = [&](size_t bytes) {
    void* p = ws + off;
    off += (bytes + 255) & ~(size_t)255;
    return p;
  };
  __bf16* xb  = (__bf16*)carve((size_t)M * D * 2);
  __bf16* wqt = (__bf16*)carve((size_t)D * D * 2);
  __bf16* wkt = (__bf16*)carve((size_t)D * D * 2);
  __bf16* wvt = (__bf16*)carve((size_t)D * D * 2);
  __bf16* wot = (__bf16*)carve((size_t)D * D * 2);
  __bf16* qb  = (__bf16*)carve((size_t)M * D * 2);
  __bf16* kb  = (__bf16*)carve((size_t)M * D * 2);
  __bf16* vt  = (__bf16*)carve((size_t)B * D * S * 2);
  float*  sc  = (float*) carve((size_t)B * S * S * 4);
  __bf16* at  = (__bf16*)carve((size_t)B * S * S * 2);
  __bf16* ob  = (__bf16*)carve((size_t)M * D * 2);
  (void)ws_size; (void)in_sizes; (void)n_in; (void)out_size;

  // 1) precision conversion + weight transposes
  {
    int n = M * D;
    cvt_bf16_kernel<<<n / 4 / 256, 256, 0, stream>>>(x, xb, n);
    dim3 tg(D / 32, D / 32), tb(32, 8);
    transpose_cvt_kernel<<<tg, tb, 0, stream>>>(Wq, wqt, D);
    transpose_cvt_kernel<<<tg, tb, 0, stream>>>(Wk, wkt, D);
    transpose_cvt_kernel<<<tg, tb, 0, stream>>>(Wv, wvt, D);
    transpose_cvt_kernel<<<tg, tb, 0, stream>>>(Wo, wot, D);
  }

  dim3 block(256);
  // 2) Q/K/V projections: (M x D) = xb (M x D) * W^T  — M=4096, N=D, K=D
  {
    dim3 grid(D / 128, M / 64, 1);
    gemm_bf16_wmma<1><<<grid, block, 0, stream>>>(xb, wqt, qb, D, D, 0, 0, 0);
    gemm_bf16_wmma<1><<<grid, block, 0, stream>>>(xb, wkt, kb, D, D, 0, 0, 0);
    gemm_bf16_wmma<2><<<grid, block, 0, stream>>>(xb, wvt, vt, D, D, 0, 0, 0);
  }
  // 3) scores = Q K^T (per batch): M=N=S, K=D. Bt = K (row-major) directly.
  {
    dim3 grid(S / 128, S / 64, B);
    gemm_bf16_wmma<0><<<grid, block, 0, stream>>>(
        qb, kb, sc, S, D, (long)S * D, (long)S * D, (long)S * S);
  }
  // 4) softmax rows (scale folded in), bf16 output
  softmax_scale_kernel<<<B * S, 256, 0, stream>>>(sc, at, scale);

  // 5) out = attn * V (per batch): M=S, N=D, K=S. Bt = Vt[b] (D x S).
  {
    dim3 grid(D / 128, S / 64, B);
    gemm_bf16_wmma<1><<<grid, block, 0, stream>>>(
        at, vt, ob, D, S, (long)S * S, (long)D * S, (long)S * D);
  }
  // 6) final projection: d_out = ob * Wo^T — M=4096, N=D, K=D, fp32 out
  {
    dim3 grid(D / 128, M / 64, 1);
    gemm_bf16_wmma<0><<<grid, block, 0, stream>>>(ob, wot, d_out, D, D, 0, 0, 0);
  }
}
